// HardNegativeMiner_20607253087043
// MI455X (gfx1250) — matrix operand order repaired and checked by reading, not compiled, not run on hardware
//
#include <hip/hip_runtime.h>
#include <hip/hip_bf16.h>
#include <stdint.h>

// ---------------------------------------------------------------------------
// Problem constants (match reference)
// ---------------------------------------------------------------------------
#define NUM_NODES 200000
#define DIM       128
#define BATCH     1024
#define KNEG      5
#define THRESH    0.7f
#define FILL_VAL  (-1.0e9f)
#define SMOOTH    0.1f

// Tiling for the WMMA similarity kernel
#define ROWS_PER_BLOCK 256            // 4 row groups cover BATCH=1024
#define COLS_PER_CHUNK 2048           // 98 chunks cover 200000 columns
#define NUM_CHUNKS     98             // ceil(200000 / 2048)
#define COLS_PER_ITER  64             // columns normalized+staged per iteration
#define NITER          (COLS_PER_CHUNK / COLS_PER_ITER)

typedef _Float16 v8h  __attribute__((ext_vector_type(8)));
typedef _Float16 v16h __attribute__((ext_vector_type(16)));
typedef float    v8f  __attribute__((ext_vector_type(8)));

// Workspace layout (bytes)
#define WS_S16_OFF   0
#define WS_PVAL_OFF  (BATCH * DIM * 2)
#define WS_PIDX_OFF  (WS_PVAL_OFF + NUM_CHUNKS * BATCH * KNEG * 4)

// LDS layout for simtopk (dynamic shared):
//   sA      : 256 rows x 128 fp16  = 65536 B (transient A staging)
//   sB      :  64 cols x 128 fp16  = 16384 B (column-major: [col][k])
//   scratch : 8 waves x 32 x 16 f32= 16384 B
//   pnorm   : 64 x 4 f32           =  1024 B
#define SMEM_SA   0
#define SMEM_SB   65536
#define SMEM_SCR  (65536 + 16384)
#define SMEM_PN   (65536 + 16384 + 16384)
#define SMEM_TOTAL (65536 + 16384 + 16384 + 1024)

// ---------------------------------------------------------------------------
// Kernel 1: gather src rows, L2-normalize, cast to fp16 -> s16 workspace
// ---------------------------------------------------------------------------
__global__ void normalize_src_kernel(const int* __restrict__ src_nodes,
                                     const float* __restrict__ memory,
                                     _Float16* __restrict__ s16) {
  __shared__ float red[DIM];
  const int b = blockIdx.x;
  const int d = threadIdx.x;
  const int node = src_nodes[b];
  const float x = memory[(size_t)node * DIM + d];
  red[d] = x * x;
  __syncthreads();
#pragma unroll
  for (int s = DIM / 2; s > 0; s >>= 1) {
    if (d < s) red[d] += red[d + s];
    __syncthreads();
  }
  const float inv = 1.0f / fmaxf(sqrtf(red[0]), 1e-12f);
  s16[(size_t)b * DIM + d] = (_Float16)(x * inv);
}

// sorted top-5 insert (descending), (val,idx) pairs in registers
#define TOP5_INSERT(v, id)                                                \
  do {                                                                    \
    if ((v) > t4v) {                                                      \
      t4v = (v); t4i = (id);                                              \
      if (t4v > t3v) { float tv = t3v; t3v = t4v; t4v = tv;               \
                       int ti = t3i; t3i = t4i; t4i = ti; }               \
      if (t3v > t2v) { float tv = t2v; t2v = t3v; t3v = tv;               \
                       int ti = t2i; t2i = t3i; t3i = ti; }               \
      if (t2v > t1v) { float tv = t1v; t1v = t2v; t2v = tv;               \
                       int ti = t1i; t1i = t2i; t2i = ti; }               \
      if (t1v > t0v) { float tv = t0v; t0v = t1v; t1v = tv;               \
                       int ti = t0i; t0i = t1i; t1i = ti; }               \
    }                                                                     \
  } while (0)

// ---------------------------------------------------------------------------
// Kernel 2: fused normalize(memory) -> fp16 GEMM (WMMA) -> masked top-5
//   - all 8 A-fragments held in registers for the whole kernel
//   - staging uses 4 threads/column (32 VGPRs) so pipeline + A-frags fit
//   - amdgpu_waves_per_eu(2,8): LDS already caps residency, so trade
//     occupancy headroom for a spill-free register budget
// ---------------------------------------------------------------------------
__global__ void __launch_bounds__(256)
__attribute__((amdgpu_waves_per_eu(2, 8)))
simtopk_kernel(const _Float16* __restrict__ s16,
               const float* __restrict__ memory,
               const int* __restrict__ dst_nodes,
               float* __restrict__ pvals,
               int* __restrict__ pidx) {
  extern __shared__ char smem[];
  _Float16* sA = (_Float16*)(smem + SMEM_SA);
  _Float16* sB = (_Float16*)(smem + SMEM_SB);
  float* scr   = (float*)(smem + SMEM_SCR);
  float* pn    = (float*)(smem + SMEM_PN);

  const int chunk = blockIdx.x;
  const int rBase = blockIdx.y * ROWS_PER_BLOCK;
  const int chunkBase = chunk * COLS_PER_CHUNK;
  const int tid  = threadIdx.x;
  const int wave = tid >> 5;
  const int lane = tid & 31;
  const int g    = lane >> 4;        // lane group 0/1
  const int n16  = lane & 15;

  // --- stage A rows into LDS once (coalesced), then hoist frags to regs ---
  {
    const uint4* srcp = (const uint4*)(s16 + (size_t)(rBase + tid) * DIM);
    uint4* dstp = (uint4*)(sA + (size_t)tid * DIM);
#pragma unroll
    for (int q = 0; q < 16; ++q) dstp[q] = srcp[q];
  }
  __syncthreads();

  // A fragments: invariant for the whole kernel (2 row-tiles x 4 K-steps)
  v16h aF[2][4];
#pragma unroll
  for (int rt = 0; rt < 2; ++rt) {
#pragma unroll
    for (int kk = 0; kk < 4; ++kk) {
      const int rL = 32 * wave + 16 * rt + n16;
      const int ka = 32 * kk + 8 * g;          // ISA 16-bit A-frag K layout
      union { v16h v; v8h h[2]; } t;
      t.h[0] = *(const v8h*)(sA + (size_t)rL * DIM + ka);
      t.h[1] = *(const v8h*)(sA + (size_t)rL * DIM + ka + 16);
      aF[rt][kk] = t.v;
    }
  }

  // each lane owns one row's top-5
  const int myRow = rBase + 32 * wave + lane;
  const int myDst = dst_nodes[myRow];
  float t0v = FILL_VAL, t1v = FILL_VAL, t2v = FILL_VAL, t3v = FILL_VAL, t4v = FILL_VAL;
  int   t0i = 0, t1i = 0, t2i = 0, t3i = 0, t4i = 0;

  float* myScr = scr + wave * (32 * 16);

  // --- software-pipelined column-group loop (64 cols/group) ---
  const int c  = tid >> 2;            // column within group (0..63)
  const int qh = tid & 3;             // quarter of the 128-dim row
  float4 xv[8];

  // prologue: load group 0
  {
    const int gcol = chunkBase + c;
    if (gcol < NUM_NODES) {
      const float4* p = (const float4*)(memory + (size_t)gcol * DIM + qh * 32);
#pragma unroll
      for (int q = 0; q < 8; ++q) xv[q] = p[q];
    } else {
#pragma unroll
      for (int q = 0; q < 8; ++q) xv[q] = make_float4(0.f, 0.f, 0.f, 0.f);
    }
  }

  for (int icg = 0; icg < NITER; ++icg) {
    const int cgBase = chunkBase + icg * COLS_PER_ITER;

    // partial sum-of-squares for my quarter row
    float ss = 0.0f;
#pragma unroll
    for (int q = 0; q < 8; ++q)
      ss += xv[q].x * xv[q].x + xv[q].y * xv[q].y +
            xv[q].z * xv[q].z + xv[q].w * xv[q].w;
    pn[c * 4 + qh] = ss;
    __syncthreads();                       // pn ready; prev WMMA done (sB free)

    const float nrm = sqrtf(pn[c * 4 + 0] + pn[c * 4 + 1] +
                            pn[c * 4 + 2] + pn[c * 4 + 3]);
    const float inv = 1.0f / fmaxf(nrm, 1e-12f);
    {
      _Float16* bdst = sB + (size_t)c * DIM + qh * 32;
#pragma unroll
      for (int q = 0; q < 8; q += 2) {
        union { _Float16 hh[8]; uint4 u; } pk;
        pk.hh[0] = (_Float16)(xv[q].x * inv);
        pk.hh[1] = (_Float16)(xv[q].y * inv);
        pk.hh[2] = (_Float16)(xv[q].z * inv);
        pk.hh[3] = (_Float16)(xv[q].w * inv);
        pk.hh[4] = (_Float16)(xv[q + 1].x * inv);
        pk.hh[5] = (_Float16)(xv[q + 1].y * inv);
        pk.hh[6] = (_Float16)(xv[q + 1].z * inv);
        pk.hh[7] = (_Float16)(xv[q + 1].w * inv);
        *(uint4*)(bdst + q * 4) = pk.u;
      }
    }

    // issue next group's global loads now; they fly during the WMMA phase
    if (icg + 1 < NITER) {
      const int gcol = cgBase + COLS_PER_ITER + c;
      if (gcol < NUM_NODES) {
        const float4* p = (const float4*)(memory + (size_t)gcol * DIM + qh * 32);
#pragma unroll
        for (int q = 0; q < 8; ++q) xv[q] = p[q];
      } else {
#pragma unroll
        for (int q = 0; q < 8; ++q) xv[q] = make_float4(0.f, 0.f, 0.f, 0.f);
      }
    }
    __syncthreads();                       // sB ready

    // --- WMMA: 2 row-tiles x 4 col-tiles; only B-frags come from LDS ---
    for (int ct = 0; ct < 4; ++ct) {
      v8f acc0 = {};
      v8f acc1 = {};
      const int cLocal = ct * 16 + n16;
#pragma unroll
      for (int kk = 0; kk < 4; ++kk) {
        const int kb = 32 * kk + 8 * g;    // ISA 16-bit B-frag K layout
        union { v16h v; v8h h[2]; } bf;
        bf.h[0] = *(const v8h*)(sB + (size_t)cLocal * DIM + kb);
        bf.h[1] = *(const v8h*)(sB + (size_t)cLocal * DIM + kb + 16);
        acc0 = __builtin_amdgcn_wmma_f32_16x16x32_f16(
            false, aF[0][kk], false, bf.v, (short)0, acc0, false, false);
        acc1 = __builtin_amdgcn_wmma_f32_16x16x32_f16(
            false, aF[1][kk], false, bf.v, (short)0, acc1, false, false);
      }

      // spill 2 x 16x16 f32 tiles into per-wave scratch: [rowInWave][col]
#pragma unroll
      for (int v = 0; v < 8; ++v) {
        myScr[(v + 8 * g) * 16 + n16]      = acc0[v];
        myScr[(16 + v + 8 * g) * 16 + n16] = acc1[v];
      }
      asm volatile("s_wait_dscnt 0x0" ::: "memory");  // cross-lane RAW in wave

      const int colBase = cgBase + ct * 16;
      const float4* rowp = (const float4*)(myScr + lane * 16);
#pragma unroll
      for (int jj = 0; jj < 4; ++jj) {
        const float4 f = rowp[jj];
        const int c0 = colBase + jj * 4;
        if (f.x > THRESH && (c0 + 0) != myDst && (c0 + 0) < NUM_NODES) TOP5_INSERT(f.x, c0 + 0);
        if (f.y > THRESH && (c0 + 1) != myDst && (c0 + 1) < NUM_NODES) TOP5_INSERT(f.y, c0 + 1);
        if (f.z > THRESH && (c0 + 2) != myDst && (c0 + 2) < NUM_NODES) TOP5_INSERT(f.z, c0 + 2);
        if (f.w > THRESH && (c0 + 3) != myDst && (c0 + 3) < NUM_NODES) TOP5_INSERT(f.w, c0 + 3);
      }
      asm volatile("s_wait_dscnt 0x0" ::: "memory");  // reads precede next stores
    }
  }

  // --- write chunk-local top-5 for my row ---
  const size_t base = ((size_t)chunk * BATCH + myRow) * KNEG;
  pvals[base + 0] = t0v; pvals[base + 1] = t1v; pvals[base + 2] = t2v;
  pvals[base + 3] = t3v; pvals[base + 4] = t4v;
  pidx[base + 0] = t0i; pidx[base + 1] = t1i; pidx[base + 2] = t2i;
  pidx[base + 3] = t3i; pidx[base + 4] = t4i;
}

// ---------------------------------------------------------------------------
// JAX threefry2x32, key = jax.random.key(42) -> (k0=0, k1=42)
// ---------------------------------------------------------------------------
__device__ __forceinline__ uint32_t rotl32(uint32_t x, int r) {
  return (x << r) | (x >> (32 - r));
}
__device__ void threefry2x32(uint32_t k0, uint32_t k1, uint32_t x0, uint32_t x1,
                             uint32_t* o0, uint32_t* o1) {
  const uint32_t k2 = k0 ^ k1 ^ 0x1BD11BDAu;
  x0 += k0; x1 += k1;
#define TF_ROUND(r) { x0 += x1; x1 = rotl32(x1, r); x1 ^= x0; }
  TF_ROUND(13) TF_ROUND(15) TF_ROUND(26) TF_ROUND(6)
  x0 += k1; x1 += k2 + 1u;
  TF_ROUND(17) TF_ROUND(29) TF_ROUND(16) TF_ROUND(24)
  x0 += k2; x1 += k0 + 2u;
  TF_ROUND(13) TF_ROUND(15) TF_ROUND(26) TF_ROUND(6)
  x0 += k0; x1 += k1 + 3u;
  TF_ROUND(17) TF_ROUND(29) TF_ROUND(16) TF_ROUND(24)
  x0 += k1; x1 += k2 + 4u;
  TF_ROUND(13) TF_ROUND(15) TF_ROUND(26) TF_ROUND(6)
  x0 += k2; x1 += k0 + 5u;
#undef TF_ROUND
  *o0 = x0; *o1 = x1;
}

__device__ __forceinline__ uint32_t jax_randbits_5120(int i) {
  const int half = (BATCH * KNEG) / 2;  // 2560
  uint32_t o0, o1;
  if (i < half) {
    threefry2x32(0u, 42u, (uint32_t)i, (uint32_t)(i + half), &o0, &o1);
    return o0;
  } else {
    threefry2x32(0u, 42u, (uint32_t)(i - half), (uint32_t)i, &o0, &o1);
    return o1;
  }
}

// ---------------------------------------------------------------------------
// Kernel 3: per-row merge of chunk-local top-5s + assemble all outputs
// out layout (floats): new_src[6144] | new_dst[6144] | new_labels[6144]
//                      | new_ts[6144] | topk_vals[5120]
// ---------------------------------------------------------------------------
__global__ void reduce_assemble_kernel(const float* __restrict__ pvals,
                                       const int* __restrict__ pidx,
                                       const int* __restrict__ src_nodes,
                                       const int* __restrict__ dst_nodes,
                                       const float* __restrict__ labels,
                                       const float* __restrict__ timestamps,
                                       float* __restrict__ out) {
  const int b = blockIdx.x * blockDim.x + threadIdx.x;
  if (b >= BATCH) return;

  float t0v = FILL_VAL, t1v = FILL_VAL, t2v = FILL_VAL, t3v = FILL_VAL, t4v = FILL_VAL;
  int   t0i = 0, t1i = 1, t2i = 2, t3i = 3, t4i = 4;
  for (int ch = 0; ch < NUM_CHUNKS; ++ch) {
    const size_t base = ((size_t)ch * BATCH + b) * KNEG;
#pragma unroll
    for (int j = 0; j < KNEG; ++j) {
      const float v = pvals[base + j];
      const int  id = pidx[base + j];
      if (v > FILL_VAL) TOP5_INSERT(v, id);
    }
  }

  const int   S = BATCH * (1 + KNEG);  // 6144 per output segment
  const int   mySrc = src_nodes[b];
  const float lab = labels[b];
  const float ts  = timestamps[b];
  const float labPos = lab * (1.0f - SMOOTH) + (1.0f - lab) * SMOOTH;
  const float labNeg = SMOOTH;

  out[0 * S + b] = (float)mySrc;
  out[1 * S + b] = (float)dst_nodes[b];
  out[2 * S + b] = labPos;
  out[3 * S + b] = ts;

  float tv[KNEG] = {t0v, t1v, t2v, t3v, t4v};
  int   ti[KNEG] = {t0i, t1i, t2i, t3i, t4i};
#pragma unroll
  for (int j = 0; j < KNEG; ++j) {
    const int i = b * KNEG + j;
    uint32_t bits = jax_randbits_5120(i);
    int rnd = (int)(bits % (uint32_t)NUM_NODES);
    if (rnd == mySrc) rnd = (rnd + 1) % NUM_NODES;
    const bool isReal = tv[j] > FILL_VAL + 1.0f;
    const int fdst = isReal ? ti[j] : rnd;

    out[0 * S + BATCH + i] = (float)mySrc;
    out[1 * S + BATCH + i] = (float)fdst;
    out[2 * S + BATCH + i] = labNeg;
    out[3 * S + BATCH + i] = ts;
    out[4 * S + i]         = tv[j];
  }
}

// ---------------------------------------------------------------------------
extern "C" void kernel_launch(void* const* d_in, const int* in_sizes, int n_in,
                              void* d_out, int out_size, void* d_ws, size_t ws_size,
                              hipStream_t stream) {
  (void)in_sizes; (void)n_in; (void)out_size; (void)ws_size;
  const int*   src_nodes  = (const int*)d_in[0];
  const int*   dst_nodes  = (const int*)d_in[1];
  const float* labels     = (const float*)d_in[2];
  const float* timestamps = (const float*)d_in[3];
  const float* memory     = (const float*)d_in[4];
  float* out = (float*)d_out;

  char* ws = (char*)d_ws;
  _Float16* s16 = (_Float16*)(ws + WS_S16_OFF);
  float* pvals  = (float*)(ws + WS_PVAL_OFF);
  int*   pidx   = (int*)(ws + WS_PIDX_OFF);

  normalize_src_kernel<<<BATCH, DIM, 0, stream>>>(src_nodes, memory, s16);

  dim3 grid(NUM_CHUNKS, BATCH / ROWS_PER_BLOCK);
  simtopk_kernel<<<grid, 256, SMEM_TOTAL, stream>>>(s16, memory, dst_nodes,
                                                    pvals, pidx);

  reduce_assemble_kernel<<<(BATCH + 255) / 256, 256, 0, stream>>>(
      pvals, pidx, src_nodes, dst_nodes, labels, timestamps, out);
}
